// PadimModel_27436251087021
// MI455X (gfx1250) — matrix operand order repaired and checked by reading, not compiled, not run on hardware
//
#include <hip/hip_runtime.h>
#include <stdint.h>

typedef __attribute__((ext_vector_type(16))) __bf16 v16bf;
typedef __attribute__((ext_vector_type(8)))  float  v8f;

#define C_IN   448
#define NPIX   4096     // 64*64 spatial locations (GEMM M)
#define DDIM   100
#define LDIM   4096     // GEMM N
#define DPAD   112      // padded row stride for f / w
#define KQUAD  5050     // packed upper-triangle d<=e
#define KLIN   5056     // start of linear-term slots
#define KCONST 5156     // constant-term slot
#define KPAD   5184     // 162 * 32
#define NKT    162      // k-tiles of 32
#define MT     256      // NPIX/16 m-tiles
#define NT     256      // LDIM/16 n-tiles
#define FRAG_DW 8       // dwords per lane per 16x32 bf16 fragment

#if __has_builtin(__builtin_amdgcn_global_load_async_to_lds_b128)
#define HAVE_ASYNC 1
typedef int v4i __attribute__((__vector_size__(4 * sizeof(int))));
typedef __attribute__((address_space(1))) v4i gv4i;   // global int4
typedef __attribute__((address_space(3))) v4i lv4i;   // LDS int4
#else
#define HAVE_ASYNC 0
#endif

static __device__ __forceinline__ unsigned short f2bf(float x){
    unsigned u = __float_as_uint(x);
    u += 0x7FFFu + ((u >> 16) & 1u);          // round-to-nearest-even
    return (unsigned short)(u >> 16);
}
// order-preserving float -> uint mapping (for atomic min on float data)
static __device__ __forceinline__ unsigned fenc(float x){
    unsigned b = __float_as_uint(x);
    return (b & 0x80000000u) ? ~b : (b | 0x80000000u);
}
static __device__ __forceinline__ float fdec(unsigned u){
    unsigned b = (u & 0x80000000u) ? (u ^ 0x80000000u) : ~u;
    return __uint_as_float(b);
}

// ---------------- init: LUT (k -> packed (d,e)) + minbuf = +inf ----------------
__global__ __launch_bounds__(256) void k_init(unsigned* __restrict__ minbuf,
                                              unsigned* __restrict__ lut){
    int t = blockIdx.x * 256 + threadIdx.x;
    if (t < NPIX) minbuf[t] = 0xFFFFFFFFu;
    if (blockIdx.x == 0 && threadIdx.x < DDIM){
        int d = threadIdx.x;
        int base = d * DDIM - (d * (d - 1)) / 2;
        for (int e = d; e < DDIM; ++e)
            lut[base + (e - d)] = (unsigned)d | ((unsigned)e << 8);
    }
}

// ---------------- projection: fmat[n][d] = sum_c feat[c][n]*pw[d][c] + pb[d] ----
__global__ __launch_bounds__(256) void k_proj(const float* __restrict__ feat,
        const float* __restrict__ pw, const float* __restrict__ pb,
        float* __restrict__ fmat){
    int n = blockIdx.x * 256 + threadIdx.x;     // coalesced over n
    int d = blockIdx.y;
    const float* f = feat + n;
    const float* w = pw + (size_t)d * C_IN;     // uniform -> scalar loads
    float acc = pb[d];
    #pragma unroll 4
    for (int c = 0; c < C_IN; ++c) acc += f[(size_t)c * NPIX] * w[c];
    fmat[(size_t)n * DPAD + d] = acc;
}

// ---------------- per-l: w = (A + A^T) m ,  const = m^T A m --------------------
__global__ __launch_bounds__(128) void k_wconst(const float* __restrict__ ic,
        const float* __restrict__ mean, float* __restrict__ wv,
        float* __restrict__ cv){
    int l = blockIdx.x;
    int d = threadIdx.x;
    __shared__ float ml[DDIM];
    __shared__ float part[DDIM];
    if (d < DDIM) ml[d] = mean[(size_t)d * LDIM + l];
    __syncthreads();
    if (d < DDIM){
        const float* A = ic + (size_t)l * DDIM * DDIM;
        float c = 0.f, r = 0.f;
        for (int e = 0; e < DDIM; ++e){
            float me = ml[e];
            c += A[d * DDIM + e] * me;   // (A m)_d
            r += A[e * DDIM + d] * me;   // (A^T m)_d
        }
        wv[(size_t)l * DPAD + d] = c + r;
        part[d] = ml[d] * c;
    }
    __syncthreads();
    if (d == 0){
        float s = 0.f;
        for (int i = 0; i < DDIM; ++i) s += part[i];
        cv[l] = s;
    }
}

// ------------- A fragments: 16x32 bf16 per (mtile,ktile), ISA lane layout -------
__global__ __launch_bounds__(256) void k_afrag(const float* __restrict__ fmat,
        const unsigned* __restrict__ lut, unsigned* __restrict__ Af){
    size_t tid = (size_t)blockIdx.x * 256 + threadIdx.x;
    int lane = (int)(tid & 31);
    size_t frag = tid >> 5;
    int kt = (int)(frag % NKT);
    int mt = (int)(frag / NKT);
    int m = mt * 16 + (lane & 15);
    int half = lane >> 4;
    const float* fr = fmat + (size_t)m * DPAD;
    unsigned out[8];
    #pragma unroll
    for (int j = 0; j < 8; ++j){
        unsigned pk = 0;
        #pragma unroll
        for (int p = 0; p < 2; ++p){
            // 16-bit A 16x32 layout: VGPR 0..3 -> K = 2j+p + 8*half,
            //                        VGPR 4..7 -> K = 16 + 2(j-4)+p + 8*half
            int ks = (j < 4 ? 2 * j : 16 + 2 * (j - 4)) + 8 * half + p;
            int k = kt * 32 + ks;
            float v;
            if (k < KQUAD){
                unsigned de = lut[k];
                v = fr[de & 255u] * fr[de >> 8];        // f_d * f_e
            } else if (k >= KLIN && k < KLIN + DDIM){
                v = fr[k - KLIN];                       // f_d (linear term)
            } else if (k == KCONST){
                v = 1.0f;                               // const term carrier
            } else {
                v = 0.0f;                               // padding
            }
            pk |= (unsigned)f2bf(v) << (16 * p);
        }
        out[j] = pk;
    }
    uint4* dst = (uint4*)(Af + tid * FRAG_DW);
    dst[0] = make_uint4(out[0], out[1], out[2], out[3]);
    dst[1] = make_uint4(out[4], out[5], out[6], out[7]);
}

// ------------- B fragments: 32x16 bf16 per (ntile,ktile) -----------------------
__global__ __launch_bounds__(256) void k_bfrag(const float* __restrict__ ic,
        const unsigned* __restrict__ lut, const float* __restrict__ wv,
        const float* __restrict__ cv, unsigned* __restrict__ Bf){
    size_t tid = (size_t)blockIdx.x * 256 + threadIdx.x;
    int lane = (int)(tid & 31);
    size_t frag = tid >> 5;
    int kt = (int)(frag % NKT);
    int nt = (int)(frag / NKT);
    int l = nt * 16 + (lane & 15);
    int kup = (lane >= 16) ? 16 : 0;
    const float* A = ic + (size_t)l * DDIM * DDIM;
    unsigned out[8];
    #pragma unroll
    for (int j = 0; j < 8; ++j){
        unsigned pk = 0;
        #pragma unroll
        for (int p = 0; p < 2; ++p){
            // 16-bit B 32x16 layout: VGPR 0..3 -> K = 2j+p + 8*half,
            //                        VGPR 4..7 -> K = 16 + 2(j-4)+p + 8*half
            int ks = (j < 4 ? 2 * j : 16 + 2 * (j - 4)) + p + (kup >> 1);
            int k = kt * 32 + ks;
            float v;
            if (k < KQUAD){
                unsigned de = lut[k];
                int d = (int)(de & 255u), e = (int)(de >> 8);
                v = (d == e) ? A[d * DDIM + d]
                             : (A[d * DDIM + e] + A[e * DDIM + d]); // symmetrized
            } else if (k >= KLIN && k < KLIN + DDIM){
                v = -wv[(size_t)l * DPAD + (k - KLIN)];             // -(A+A^T)m
            } else if (k == KCONST){
                v = cv[l];                                          // m^T A m
            } else {
                v = 0.0f;
            }
            pk |= (unsigned)f2bf(v) << (16 * p);
        }
        out[j] = pk;
    }
    uint4* dst = (uint4*)(Bf + tid * FRAG_DW);
    dst[0] = make_uint4(out[0], out[1], out[2], out[3]);
    dst[1] = make_uint4(out[4], out[5], out[6], out[7]);
}

// ------------- GEMM (dist = ff@Bsym - f@w + const) fused with min over L -------
// WG tile 128x128, 8 waves (2 M x 4 N), per-wave 64x32 (4x2 WMMA tiles).
// Per k-step the WG stages 16 fragments (8 A + 8 B = 16 KB) into LDS with
// async global->LDS copies (double buffered), then all waves read from LDS.
__global__ __launch_bounds__(256) void k_gemm_min(const unsigned* __restrict__ Af,
        const unsigned* __restrict__ Bf, unsigned* __restrict__ minbuf){
    __shared__ __align__(32) unsigned stage[2][4096];   // 2 x 16 KB
    __shared__ unsigned rowmin[128];

    int lane = threadIdx.x & 31;
    int wave = threadIdx.x >> 5;                 // 8 waves: 2 (M) x 4 (N)
    int mt0w = blockIdx.y * 8;                   // WG m-tile base
    int nt0w = blockIdx.x * 8;                   // WG n-tile base
    int wm = (wave & 1) * 4;                     // wave m-subtile base (0/4)
    int wn = (wave >> 1) * 2;                    // wave n-subtile base (0/2/4/6)

    // per-thread staging assignment: 4 x 16B units out of 1024 per k-step
    // unit -> (chunk, pos): chunk 0..7 = A frag, 8..15 = B frag; pos 0..63
    const unsigned* gsrc[4];
    #pragma unroll
    for (int i = 0; i < 4; ++i){
        int unit  = threadIdx.x + i * 256;
        int chunk = unit >> 6;
        int pos   = unit & 63;
        gsrc[i] = (chunk < 8)
            ? Af + (((size_t)(mt0w + chunk)     * NKT) * 256 + (size_t)pos * 4)
            : Bf + (((size_t)(nt0w + chunk - 8) * NKT) * 256 + (size_t)pos * 4);
    }

    auto issue = [&](int buf){
        #pragma unroll
        for (int i = 0; i < 4; ++i){
            int unit = threadIdx.x + i * 256;
            unsigned* l = &stage[buf][unit * 4];
#if HAVE_ASYNC
            __builtin_amdgcn_global_load_async_to_lds_b128(
                (gv4i*)gsrc[i], (lv4i*)l, 0, 0);
#else
            *(uint4*)l = *(const uint4*)gsrc[i];
#endif
            gsrc[i] += 256;                      // advance one k-step (1 KB)
        }
    };

    v8f z = {0.f, 0.f, 0.f, 0.f, 0.f, 0.f, 0.f, 0.f};
    v8f acc[4][2];
    #pragma unroll
    for (int mi = 0; mi < 4; ++mi)
        #pragma unroll
        for (int ni = 0; ni < 2; ++ni) acc[mi][ni] = z;

    if (threadIdx.x < 128) rowmin[threadIdx.x] = 0xFFFFFFFFu;

    issue(0);                                    // prologue fill, k-step 0
    for (int kt = 0; kt < NKT; ++kt){
        int buf = kt & 1;
#if HAVE_ASYNC
#if __has_builtin(__builtin_amdgcn_s_wait_asynccnt)
        __builtin_amdgcn_s_wait_asynccnt(0);
#else
        asm volatile("s_wait_asynccnt 0" ::: "memory");
#endif
#endif
        __syncthreads();                         // tile kt ready; kt-1 consumed
        if (kt + 1 < NKT) issue(buf ^ 1);        // overlap fill of kt+1

        const unsigned* As = &stage[buf][0];
        v16bf a[4], b[2];
        #pragma unroll
        for (int mi = 0; mi < 4; ++mi)
            a[mi] = *(const v16bf*)(As + (size_t)(wm + mi) * 256 + (size_t)lane * 8);
        #pragma unroll
        for (int ni = 0; ni < 2; ++ni)
            b[ni] = *(const v16bf*)(As + 2048 + (size_t)(wn + ni) * 256 + (size_t)lane * 8);
        #pragma unroll
        for (int mi = 0; mi < 4; ++mi)
            #pragma unroll
            for (int ni = 0; ni < 2; ++ni)
                acc[mi][ni] = __builtin_amdgcn_wmma_f32_16x16x32_bf16(
                    false, a[mi], false, b[ni], (short)0, acc[mi][ni],
                    false, false);
    }

    // min over the 128 columns handled by this WG, then global atomic min
    __syncthreads();
    #pragma unroll
    for (int mi = 0; mi < 4; ++mi){
        int rbase = (wave & 1) * 64 + mi * 16 + ((lane < 16) ? 0 : 8);
        #pragma unroll
        for (int r = 0; r < 8; ++r){
            float v = fminf(acc[mi][0][r], acc[mi][1][r]);
            atomicMin(&rowmin[rbase + r], fenc(v));
        }
    }
    __syncthreads();
    if (threadIdx.x < 128)
        atomicMin(minbuf + (size_t)blockIdx.y * 128 + threadIdx.x,
                  rowmin[threadIdx.x]);
}

// ------------- decode min map + global min/max reduction -----------------------
__global__ __launch_bounds__(256) void k_reduce(const unsigned* __restrict__ minbuf,
        float* __restrict__ minf, float* __restrict__ gmm){
    __shared__ float smin[256], smax[256];
    int t = threadIdx.x;
    float lmin = 3.4e38f, lmax = -3.4e38f;
    for (int i = t; i < NPIX; i += 256){
        float f = fdec(minbuf[i]);
        minf[i] = f;
        lmin = fminf(lmin, f);
        lmax = fmaxf(lmax, f);
    }
    smin[t] = lmin; smax[t] = lmax;
    __syncthreads();
    for (int s = 128; s > 0; s >>= 1){
        if (t < s){
            smin[t] = fminf(smin[t], smin[t + s]);
            smax[t] = fmaxf(smax[t], smax[t + s]);
        }
        __syncthreads();
    }
    if (t == 0){ gmm[0] = smin[0]; gmm[1] = smax[0]; }
}

// ------------- normalize + half-pixel bilinear 64 -> 256 -----------------------
__global__ __launch_bounds__(256) void k_resize(const float* __restrict__ minf,
        const float* __restrict__ gmm, float* __restrict__ out){
    int idx = blockIdx.x * 256 + threadIdx.x;   // 65536 outputs
    int x = idx & 255, y = idx >> 8;
    float gmin = gmm[0];
    float inv = 1.0f / (gmm[1] - gmin + 1e-8f);
    float sx = (x + 0.5f) * 0.25f - 0.5f;
    float sy = (y + 0.5f) * 0.25f - 0.5f;
    float fxf = floorf(sx), fyf = floorf(sy);
    int x0 = (int)fxf, y0 = (int)fyf;
    float wx = sx - fxf, wy = sy - fyf;
    int x0c = min(max(x0, 0), 63), x1c = min(max(x0 + 1, 0), 63);
    int y0c = min(max(y0, 0), 63), y1c = min(max(y0 + 1, 0), 63);
    float v00 = minf[y0c * 64 + x0c], v01 = minf[y0c * 64 + x1c];
    float v10 = minf[y1c * 64 + x0c], v11 = minf[y1c * 64 + x1c];
    float v = v00 * (1.f - wx) * (1.f - wy) + v01 * wx * (1.f - wy)
            + v10 * (1.f - wx) * wy        + v11 * wx * wy;
    out[idx] = (v - gmin) * inv;
}

static inline size_t alignup(size_t x){ return (x + 255) & ~(size_t)255; }

extern "C" void kernel_launch(void* const* d_in, const int* in_sizes, int n_in,
                              void* d_out, int out_size, void* d_ws, size_t ws_size,
                              hipStream_t stream) {
    (void)in_sizes; (void)n_in; (void)out_size; (void)ws_size;
    const float* feat = (const float*)d_in[0];   // (1,448,64,64)
    const float* pw   = (const float*)d_in[1];   // (100,448)
    const float* pb   = (const float*)d_in[2];   // (100,)
    const float* mean = (const float*)d_in[3];   // (100,4096)
    const float* ic   = (const float*)d_in[4];   // (4096,100,100)
    float* out = (float*)d_out;                  // (1,1,256,256)

    char* ws = (char*)d_ws;
    size_t off = 0;
    auto take = [&](size_t bytes){ size_t o = off; off += alignup(bytes); return o; };
    float*    fmat   = (float*)   (ws + take((size_t)NPIX * DPAD * 4));
    float*    wv     = (float*)   (ws + take((size_t)LDIM * DPAD * 4));
    float*    cv     = (float*)   (ws + take((size_t)LDIM * 4));
    unsigned* lut    = (unsigned*)(ws + take((size_t)KLIN * 4));
    unsigned* minbuf = (unsigned*)(ws + take((size_t)NPIX * 4));
    float*    minf   = (float*)   (ws + take((size_t)NPIX * 4));
    float*    gmm    = (float*)   (ws + take(256));
    unsigned* Af     = (unsigned*)(ws + take((size_t)MT * NKT * 32 * FRAG_DW * 4));
    unsigned* Bf     = (unsigned*)(ws + take((size_t)NT * NKT * 32 * FRAG_DW * 4));

    k_init  <<<16, 256, 0, stream>>>(minbuf, lut);
    k_proj  <<<dim3(16, DDIM), 256, 0, stream>>>(feat, pw, pb, fmat);
    k_wconst<<<LDIM, 128, 0, stream>>>(ic, mean, wv, cv);
    k_afrag <<<(MT * NKT * 32) / 256, 256, 0, stream>>>(fmat, lut, Af);
    k_bfrag <<<(NT * NKT * 32) / 256, 256, 0, stream>>>(ic, lut, wv, cv, Bf);
    k_gemm_min<<<dim3(32, 32), 256, 0, stream>>>(Af, Bf, minbuf);
    k_reduce<<<1, 256, 0, stream>>>(minbuf, minf, gmm);
    k_resize<<<256, 256, 0, stream>>>(minf, gmm, out);
}
